// GIN_76484777607240
// MI455X (gfx1250) — matrix operand order repaired and checked by reading, not compiled, not run on hardware
//
#include <hip/hip_runtime.h>

// ---------------------------------------------------------------------------
// GIN forward for MI455X (gfx1250). wave32, WMMA bf16 GEMMs with LDS-staged
// weights (async-to-LDS path), HW f32 atomics for graph scatter/pool.
// ---------------------------------------------------------------------------

typedef __attribute__((ext_vector_type(16))) __bf16 v16bf;
typedef __attribute__((ext_vector_type(8)))  float  v8f;
typedef int v4i_b128 __attribute__((vector_size(16)));   // async b128 payload

#define ATOMIC_FADD(p, v) \
  __hip_atomic_fetch_add((p), (v), __ATOMIC_RELAXED, __HIP_MEMORY_SCOPE_AGENT)

__device__ __forceinline__ unsigned short f2bf(float f) {
  unsigned int u = __float_as_uint(f);
  u += 0x7FFFu + ((u >> 16) & 1u);   // round-to-nearest-even
  return (unsigned short)(u >> 16);
}

union BFrag { v16bf v; uint4 u[2]; };

// --------------------------- edge scatter-add ------------------------------
// One wave per edge: 32 lanes x float2 = 64 features. agg[dst] += h[src].
__global__ __launch_bounds__(256) void scatter_add_k(
    const int* __restrict__ src, const int* __restrict__ dst,
    const float* __restrict__ h, float* __restrict__ agg, int E) {
  int t = blockIdx.x * 256 + threadIdx.x;
  int e = t >> 5;
  if (e >= E) return;
  int lane = t & 31;
  long s = src[e], d = dst[e];
  float2 v = ((const float2*)(h + s * 64))[lane];
  float* a = agg + d * 64 + lane * 2;
  ATOMIC_FADD(a + 0, v.x);
  ATOMIC_FADD(a + 1, v.y);
}

// --------------------- z = agg + h, convert to bf16 ------------------------
__global__ __launch_bounds__(256) void prep_bf16_k(
    const float* __restrict__ agg, const float* __restrict__ h,
    unsigned short* __restrict__ out) {
  size_t i = (size_t)blockIdx.x * 256 + threadIdx.x;  // over nelem/4
  float4 a = ((const float4*)agg)[i];
  float4 b = ((const float4*)h)[i];
  ushort4 r;
  r.x = f2bf(a.x + b.x);
  r.y = f2bf(a.y + b.y);
  r.z = f2bf(a.z + b.z);
  r.w = f2bf(a.w + b.w);
  ((ushort4*)out)[i] = r;
}

// ----------------- weight transpose + bf16 convert (64x64) -----------------
// W[fin][fout] f32 row-major  ->  Wt[fout][fin] bf16 (column-major for B frag)
__global__ __launch_bounds__(256) void wprep_k(
    const float* __restrict__ W, unsigned short* __restrict__ Wt) {
  int t = blockIdx.x * 256 + threadIdx.x;  // 4096
  int fin = t >> 6, fout = t & 63;
  Wt[fout * 64 + fin] = f2bf(W[t]);
}

// ------------------ GEMM: C[rows x 64] = A @ W + bias ----------------------
// 256 threads = 8 waves; block covers 128 rows. B (8 KB) staged into LDS once
// per block via gfx1250 async-to-LDS; each wave computes its full 16x64 row
// strip: A fragments loaded once, 4 column tiles x 2 chained
// v_wmma_f32_16x16x32_bf16 (K=64) = 8 WMMAs/wave, B served by ds_load_b128.
// Epilogue fuses bias add and BN column sum / sum-of-squares. The row guard
// is hoisted to a wave-uniform branch so only the single tail block pays the
// exec-mask dance; all full tiles store via one unguarded s_clause.
__global__ __launch_bounds__(256) void gemm64_bn_k(
    const unsigned short* __restrict__ A,   // [rowsPad x 64] bf16 row-major
    const unsigned short* __restrict__ Wt,  // [64 x 64] bf16 col-major (n-major)
    const float* __restrict__ bias,
    float* __restrict__ C,                  // [rows x 64] f32
    float* __restrict__ colsum, float* __restrict__ colsumsq, int nrows) {
  __shared__ unsigned short sB[64 * 64];    // 8 KB

  // ---- stage B into LDS: 256 threads x 32 B ----
  {
    int t = threadIdx.x;
    const char* gsrc = (const char*)Wt + t * 32;
    char* ldst = (char*)sB + t * 32;
#if defined(__gfx1250__) && __has_builtin(__builtin_amdgcn_global_load_async_to_lds_b128)
    __attribute__((address_space(1))) v4i_b128* gp =
        (__attribute__((address_space(1))) v4i_b128*)(void*)gsrc;
    __attribute__((address_space(3))) v4i_b128* lp =
        (__attribute__((address_space(3))) v4i_b128*)(void*)ldst;
    __builtin_amdgcn_global_load_async_to_lds_b128(gp, lp, 0, 0);
    __builtin_amdgcn_global_load_async_to_lds_b128(gp, lp, 16, 0);
#if __has_builtin(__builtin_amdgcn_s_wait_asynccnt)
    __builtin_amdgcn_s_wait_asynccnt(0);
#else
    asm volatile("s_wait_asynccnt 0x0" ::: "memory");
#endif
#else
    ((uint4*)ldst)[0] = ((const uint4*)gsrc)[0];
    ((uint4*)ldst)[1] = ((const uint4*)gsrc)[1];
#endif
  }
  __syncthreads();

  int wave  = threadIdx.x >> 5;
  int lane  = threadIdx.x & 31;
  int lhalf = lane >> 4;       // 0 | 1
  int l15   = lane & 15;
  int rowTile = blockIdx.x * 128 + wave * 16;

  // A fragments: per ISA 16-bit A layout, lane row = rowTile + l15,
  // K = {kofs..kofs+7, kofs+16..kofs+23}, kofs = lhalf*8.
  const unsigned short* arow = A + (size_t)(rowTile + l15) * 64;
  int kofs = lhalf * 8;
  BFrag a0, a1;
  a0.u[0] = *(const uint4*)(arow + kofs);
  a0.u[1] = *(const uint4*)(arow + kofs + 16);
  a1.u[0] = *(const uint4*)(arow + 32 + kofs);
  a1.u[1] = *(const uint4*)(arow + 32 + kofs + 16);

  const bool fullTile = (rowTile + 16 <= nrows);   // wave-uniform

#pragma unroll
  for (int ct = 0; ct < 4; ++ct) {
    int colTile = ct * 16;
    // B fragments: lane column = colTile + l15, K = {kofs..kofs+7, kofs+16..kofs+23}, kofs = lhalf*8.
    const unsigned short* bcol = sB + (colTile + l15) * 64 + lhalf * 8;
    BFrag b0, b1;
    b0.u[0] = *(const uint4*)(bcol + 0);
    b0.u[1] = *(const uint4*)(bcol + 16);
    b1.u[0] = *(const uint4*)(bcol + 32);
    b1.u[1] = *(const uint4*)(bcol + 48);

    v8f c = {};
    c = __builtin_amdgcn_wmma_f32_16x16x32_bf16(false, a0.v, false, b0.v,
                                                (short)0, c, false, false);
    c = __builtin_amdgcn_wmma_f32_16x16x32_bf16(false, a1.v, false, b1.v,
                                                (short)0, c, false, false);

    int gcol = colTile + l15;
    float bs = bias[gcol];
    float s = 0.f, sq = 0.f;
    float* crow = C + (size_t)(rowTile + lhalf * 8) * 64 + gcol;
    if (fullTile) {
      // Hot path: unguarded 8-store clause (C row = rowTile + lhalf*8 + v).
#pragma unroll
      for (int v = 0; v < 8; ++v) {
        float val = c[v] + bs;
        crow[(size_t)v * 64] = val;
        s  += val;
        sq += val * val;
      }
    } else {
      // Tail block only (one per GEMM): per-row guard.
#pragma unroll
      for (int v = 0; v < 8; ++v) {
        int row = rowTile + lhalf * 8 + v;
        if (row < nrows) {
          float val = c[v] + bs;
          crow[(size_t)v * 64] = val;
          s  += val;
          sq += val * val;
        }
      }
    }
    ATOMIC_FADD(&colsum[gcol], s);
    ATOMIC_FADD(&colsumsq[gcol], sq);
  }
}

// ------------- fold BN stats into per-feature scale / shift ----------------
__global__ void bn_stats_k(const float* __restrict__ colsum,
                           const float* __restrict__ colsumsq,
                           const float* __restrict__ gamma,
                           const float* __restrict__ beta,
                           float* __restrict__ scale, float* __restrict__ shift,
                           float invN) {
  int j = threadIdx.x;  // 64
  float m   = colsum[j] * invN;
  float var = colsumsq[j] * invN - m * m;
  float sc  = gamma[j] * rsqrtf(var + 1e-5f);
  scale[j] = sc;
  shift[j] = beta[j] - sc * m;
}

// ------------------- BN apply + ReLU, multi-format out ---------------------
__global__ __launch_bounds__(256) void bn_relu_k(
    const float* __restrict__ z, const float* __restrict__ scale,
    const float* __restrict__ shift, unsigned short* __restrict__ out_bf,
    float* __restrict__ out_f, float* __restrict__ xs_slice) {
  size_t i = (size_t)blockIdx.x * 256 + threadIdx.x;  // over N*64
  int col = (int)(i & 63);
  float y = fmaxf(scale[col] * z[i] + shift[col], 0.f);
  if (out_bf)   out_bf[i] = f2bf(y);
  if (out_f)    out_f[i] = y;
  if (xs_slice) xs_slice[(i >> 6) * 256 + col] = y;
}

// --------------------- segment-sum pooling (256-wide) ----------------------
__global__ __launch_bounds__(256) void pool_k(const float* __restrict__ xs,
                                              const int* __restrict__ batch,
                                              float* __restrict__ pooled) {
  int node = blockIdx.x;
  int f = threadIdx.x;
  int g = batch[node];
  ATOMIC_FADD(&pooled[(size_t)g * 256 + f], xs[(size_t)node * 256 + f]);
}

// -------------------------- final tiny MLP ---------------------------------
__global__ __launch_bounds__(64) void mlp_k(
    const float* __restrict__ pooled, const float* __restrict__ W1,
    const float* __restrict__ b1, const float* __restrict__ W2,
    const float* __restrict__ b2, float* __restrict__ out) {
  __shared__ float hbuf[64];
  int g = blockIdx.x, j = threadIdx.x;
  const float* p = pooled + (size_t)g * 256;
  float acc = b1[j];
  for (int k = 0; k < 256; ++k) acc += p[k] * W1[k * 64 + j];
  hbuf[j] = fmaxf(acc, 0.f);
  __syncthreads();
  if (j < 10) {
    float o = b2[j];
    for (int k = 0; k < 64; ++k) o += hbuf[k] * W2[k * 10 + j];
    out[g * 10 + j] = o;
  }
}

// ---------------------------------------------------------------------------
extern "C" void kernel_launch(void* const* d_in, const int* in_sizes, int n_in,
                              void* d_out, int out_size, void* d_ws,
                              size_t ws_size, hipStream_t stream) {
  const float* x     = (const float*)d_in[0];
  const int*   ei    = (const int*)d_in[1];
  const int*   batch = (const int*)d_in[2];
  const int N = in_sizes[2];
  const int E = in_sizes[1] / 2;
  const int L = 4, G = 128;
  const int* srcp = ei;
  const int* dstp = ei + E;
  const float* lin1_W = (const float*)d_in[3 + 8 * L + 0];
  const float* lin1_b = (const float*)d_in[3 + 8 * L + 1];
  const float* lin2_W = (const float*)d_in[3 + 8 * L + 2];
  const float* lin2_b = (const float*)d_in[3 + 8 * L + 3];

  // ---- workspace layout ----
  char* ws = (char*)d_ws;
  size_t o = 0;
  auto alloc = [&](size_t bytes) -> void* {
    void* p = ws + o;
    o += (bytes + 255) & ~(size_t)255;
    return p;
  };
  const size_t nelem   = (size_t)N * 64;
  const size_t rowsPad = ((size_t)N + 127) & ~(size_t)127;  // GEMM row padding
  float*          agg    = (float*)alloc(nelem * 4);            // reused as z_pre
  float*          hbuf   = (float*)alloc(nelem * 4);
  unsigned short* gin    = (unsigned short*)alloc(rowsPad * 64 * 2); // bf16 input
  float*          xs     = (float*)alloc((size_t)N * 256 * 4);
  unsigned short* wts    = (unsigned short*)alloc(8 * 4096 * 2);
  float*          colsum   = (float*)alloc(64 * 4);
  float*          colsumsq = (float*)alloc(64 * 4);
  float*          scale    = (float*)alloc(64 * 4);
  float*          shift    = (float*)alloc(64 * 4);
  float*          pooled   = (float*)alloc((size_t)G * 256 * 4);

  // ---- weight prep: transpose + bf16 once per call ----
  for (int l = 0; l < L; ++l) {
    const float* W1 = (const float*)d_in[3 + 8 * l + 0];
    const float* W2 = (const float*)d_in[3 + 8 * l + 4];
    wprep_k<<<16, 256, 0, stream>>>(W1, wts + (size_t)(2 * l) * 4096);
    wprep_k<<<16, 256, 0, stream>>>(W2, wts + (size_t)(2 * l + 1) * 4096);
  }

  const float invN = 1.0f / (float)N;
  const int scatterBlocks = (int)(((long)E * 32 + 255) / 256);
  const int prepBlocks    = (int)(nelem / 4 / 256);
  const int gemmBlocks    = (N + 127) / 128;   // 128 rows per block, row-guarded
  const int elemBlocks    = (int)(nelem / 256);

  const float* h = x;
  for (int l = 0; l < L; ++l) {
    const float* b1  = (const float*)d_in[3 + 8 * l + 1];
    const float* g1  = (const float*)d_in[3 + 8 * l + 2];
    const float* be1 = (const float*)d_in[3 + 8 * l + 3];
    const float* b2  = (const float*)d_in[3 + 8 * l + 5];
    const float* g2  = (const float*)d_in[3 + 8 * l + 6];
    const float* be2 = (const float*)d_in[3 + 8 * l + 7];

    // agg = segment_sum(h[src], dst)
    (void)hipMemsetAsync(agg, 0, nelem * 4, stream);
    scatter_add_k<<<scatterBlocks, 256, 0, stream>>>(srcp, dstp, h, agg, E);
    // gin = bf16(agg + h)
    prep_bf16_k<<<prepBlocks, 256, 0, stream>>>(agg, h, gin);

    // z_pre = gin @ W1 + b1  (+ BN stats); reuse agg as z_pre
    (void)hipMemsetAsync(colsum, 0, 256, stream);
    (void)hipMemsetAsync(colsumsq, 0, 256, stream);
    gemm64_bn_k<<<gemmBlocks, 256, 0, stream>>>(
        gin, wts + (size_t)(2 * l) * 4096, b1, agg, colsum, colsumsq, N);
    bn_stats_k<<<1, 64, 0, stream>>>(colsum, colsumsq, g1, be1, scale, shift, invN);
    bn_relu_k<<<elemBlocks, 256, 0, stream>>>(agg, scale, shift, gin,
                                              nullptr, nullptr);

    // z_pre = gin @ W2 + b2  (+ BN stats)
    (void)hipMemsetAsync(colsum, 0, 256, stream);
    (void)hipMemsetAsync(colsumsq, 0, 256, stream);
    gemm64_bn_k<<<gemmBlocks, 256, 0, stream>>>(
        gin, wts + (size_t)(2 * l + 1) * 4096, b2, agg, colsum, colsumsq, N);
    bn_stats_k<<<1, 64, 0, stream>>>(colsum, colsumsq, g2, be2, scale, shift, invN);
    // h_next (f32) + concat slice xs[:, l*64 : l*64+64]
    bn_relu_k<<<elemBlocks, 256, 0, stream>>>(agg, scale, shift, nullptr,
                                              hbuf, xs + l * 64);
    h = hbuf;
  }

  // pooled[G x 256] = segment_sum(xs, batch)
  (void)hipMemsetAsync(pooled, 0, (size_t)G * 256 * 4, stream);
  pool_k<<<N, 256, 0, stream>>>(xs, batch, pooled);

  // out = relu(pooled @ lin1 + b) @ lin2 + b
  mlp_k<<<G, 64, 0, stream>>>(pooled, lin1_W, lin1_b, lin2_W, lin2_b,
                              (float*)d_out);
}
